// SphericalCNN_68839735820492
// MI455X (gfx1250) — matrix operand order repaired and checked by reading, not compiled, not run on hardware
//
#include <hip/hip_runtime.h>
#include <math.h>

// ---------------------------------------------------------------------------
// Types / helpers
// ---------------------------------------------------------------------------
typedef __attribute__((ext_vector_type(16))) __bf16 v16bf;
typedef __attribute__((ext_vector_type(8)))  float  v8f;

union Frag {
    v16bf v;
    unsigned short h[16];
    unsigned int   w[8];
};

__device__ __forceinline__ unsigned short f2bf(float f) {
    unsigned int u = __float_as_uint(f);
    unsigned int r = u + 0x7FFFu + ((u >> 16) & 1u);   // round-to-nearest-even
    return (unsigned short)(r >> 16);
}

// Load one WMMA A-layout fragment: two contiguous 16B segments per lane.
__device__ __forceinline__ void ldA(Frag& f, const unsigned short* p0,
                                    const unsigned short* p1) {
    uint4 x = *(const uint4*)p0;
    uint4 y = *(const uint4*)p1;
    f.w[0] = x.x; f.w[1] = x.y; f.w[2] = x.z; f.w[3] = x.w;
    f.w[4] = y.x; f.w[5] = y.y; f.w[6] = y.z; f.w[7] = y.w;
}
// Load one WMMA B-layout fragment: one contiguous 32B segment per lane.
__device__ __forceinline__ void ldB(Frag& f, const unsigned short* p) {
    uint4 x = *(const uint4*)p;
    uint4 y = *(const uint4*)(p + 8);
    f.w[0] = x.x; f.w[1] = x.y; f.w[2] = x.z; f.w[3] = x.w;
    f.w[4] = y.x; f.w[5] = y.y; f.w[6] = y.z; f.w[7] = y.w;
}

#define SO3 455
#define S2  49
#define NGRID 6912
#define NB 32

// ---------------------------------------------------------------------------
// psi0[c,g,i] = (1/sqrt(24)) sum_n Y[n,i] * w_s2[c,g,n]
// ---------------------------------------------------------------------------
__global__ void k_psi0(const float* __restrict__ Y, const float* __restrict__ w,
                       float* __restrict__ psi0) {
    int idx = blockIdx.x * blockDim.x + threadIdx.x;
    if (idx >= 128 * 128 * S2) return;
    int i = idx % S2;
    int cg = idx / S2;
    float acc = 0.f;
    for (int n = 0; n < 24; ++n)
        acc += Y[n * S2 + i] * w[cg * 24 + n];
    psi0[idx] = acc * 0.2041241452319315f;  // 1/sqrt(24)
}

// ---------------------------------------------------------------------------
// h[b,g, off+u*d+m] = (1/sqrt(128)) sum_c psi0[c,g,l2+u] * x[b,c,l2+m]
// ---------------------------------------------------------------------------
__global__ void k_s2_so3(const float* __restrict__ x, const float* __restrict__ psi0,
                         float* __restrict__ h, int d, int off, int l2) {
    int idx = blockIdx.x * blockDim.x + threadIdx.x;
    if (idx >= NB * 128 * d * d) return;
    int m = idx % d;
    int t = idx / d;
    int u = t % d;  t /= d;
    int g = t % 128;
    int b = t / 128;
    float acc = 0.f;
    for (int c = 0; c < 128; ++c)
        acc += psi0[(c * 128 + g) * S2 + l2 + u] * x[(b * 128 + c) * S2 + l2 + m];
    h[(b * 128 + g) * SO3 + off + u * d + m] = acc * 0.08838834764831845f; // 1/sqrt(128)
}

// ---------------------------------------------------------------------------
// Dg -> bf16 in two layouts: DgNK[6912][512] row-major (K padded to 512),
// DgT[512][6912] column-major.  7MB each, L2-resident.
// ---------------------------------------------------------------------------
__global__ void k_prep_dg(const float* __restrict__ Dg,
                          unsigned short* __restrict__ DgNK,
                          unsigned short* __restrict__ DgT) {
    int idx = blockIdx.x * blockDim.x + threadIdx.x;
    if (idx >= NGRID * 512) return;
    int n = idx >> 9, k = idx & 511;
    unsigned short bf = (k < SO3) ? f2bf(Dg[n * SO3 + k]) : (unsigned short)0;
    DgNK[idx] = bf;
    DgT[(size_t)k * NGRID + n] = bf;
}

// ---------------------------------------------------------------------------
// Fused SO(3) activation:  out = (Dg^T relu(Dg h^T))^T / NGRID
// One workgroup = 16 rows of h, all 455 output cols.
// n-loop over 6912 grid points in blocks of 128; within a block each wave owns
// one complete 16x16 F subtile (no cross-wave reduction), then all waves
// accumulate their 4 output tiles over the 128-point K range.
// ---------------------------------------------------------------------------
__global__ __launch_bounds__(256) void k_act(const float* __restrict__ hin,
                                             const unsigned short* __restrict__ DgNK,
                                             const unsigned short* __restrict__ DgT,
                                             float* __restrict__ hout) {
    __shared__ unsigned short sH[16][512];     // H tile, bf16, K padded to 512
    __shared__ unsigned short sF[16][128];     // relu'd F block, bf16

    const int tid  = threadIdx.x;
    const int lane = tid & 31;
    const int wv   = tid >> 5;
    const int la   = lane & 15;
    const int lg   = lane >> 4;
    const int r0   = blockIdx.x * 16;

    for (int idx = tid; idx < 16 * 512; idx += 256) {
        int m = idx >> 9, k = idx & 511;
        sH[m][k] = (k < SO3) ? f2bf(hin[(r0 + m) * SO3 + k]) : (unsigned short)0;
    }

    v8f acc0 = {}, acc1 = {}, acc2 = {}, acc3 = {};

    for (int blk = 0; blk < NGRID / 128; ++blk) {
        const int n0 = blk * 128;
        // prefetch next block's Dg chunks into cache while we compute
        if (blk + 1 < NGRID / 128) {
            __builtin_prefetch(DgNK + (size_t)(n0 + 128) * 512 + (size_t)tid * 256, 0, 1);
            __builtin_prefetch(DgT + (size_t)(tid * 2) * NGRID + n0 + 128, 0, 1);
            __builtin_prefetch(DgT + (size_t)(tid * 2 + 1) * NGRID + n0 + 128, 0, 1);
        }
        __syncthreads();   // sF (and on blk 0, sH) ready for this block
        // ---- phase 1: F[:, 16*wv .. 16*wv+15] over full K=512 ----
        {
            v8f p = {};
            const unsigned short* hrow = sH[la];
            const unsigned short* brow = DgNK + (size_t)(n0 + 16 * wv + la) * 512;
            for (int kc = 0; kc < 16; ++kc) {
                const int k0 = 32 * kc;
                Frag a, b;
                ldA(a, &hrow[k0 + 16 * lg], &hrow[k0 + 16 * lg + 8]);
                ldB(b, &brow[k0 + 16 * lg]);
                p = __builtin_amdgcn_wmma_f32_16x16x32_bf16(false, a.v, false, b.v,
                                                            (short)0, p, false, false);
            }
            // relu + 1/NGRID + quantize straight from the accumulator
#pragma unroll
            for (int v = 0; v < 8; ++v)
                sF[v + 8 * lg][16 * wv + la] =
                    f2bf(fmaxf(p[v], 0.f) * (1.0f / (float)NGRID));
        }
        __syncthreads();
        // ---- phase 2: OUT(tiles of this wave) += F x Dg_chunk (K = 128) ----
        for (int kc = 0; kc < 4; ++kc) {
            const int k0 = 32 * kc;
            Frag a;
            ldA(a, &sF[la][k0 + 16 * lg], &sF[la][k0 + 16 * lg + 8]);
#pragma unroll
            for (int t4 = 0; t4 < 4; ++t4) {
                int cb = (4 * wv + t4) * 16;
                Frag b;
                ldB(b, DgT + (size_t)(cb + la) * NGRID + n0 + k0 + 16 * lg);
                v8f& acc = (t4 == 0) ? acc0 : (t4 == 1) ? acc1 : (t4 == 2) ? acc2 : acc3;
                acc = __builtin_amdgcn_wmma_f32_16x16x32_bf16(false, a.v, false, b.v,
                                                              (short)0, acc, false, false);
            }
        }
    }
    // write out (1/NGRID already folded into F)
#pragma unroll
    for (int t4 = 0; t4 < 4; ++t4) {
        v8f acc = (t4 == 0) ? acc0 : (t4 == 1) ? acc1 : (t4 == 2) ? acc2 : acc3;
        int col = (4 * wv + t4) * 16 + la;
        if (col < SO3) {
#pragma unroll
            for (int v = 0; v < 8; ++v)
                hout[(r0 + v + 8 * lg) * SO3 + col] = acc[v];
        }
    }
}

// ---------------------------------------------------------------------------
// psi_l packed as WMMA A operand: A[(g*d+v)*K + c*d+u] (K = ci*d), bf16,
// with both 1/sqrt(72) and 1/sqrt(f_in*d) folded in via `inv`.
// ---------------------------------------------------------------------------
__global__ void k_psi_l(const float* __restrict__ D, const float* __restrict__ w,
                        unsigned short* __restrict__ A, int ci, int co, int d,
                        int off, float inv) {
    int K = ci * d;
    long long total = (long long)co * d * K;
    long long idx = (long long)blockIdx.x * blockDim.x + threadIdx.x;
    if (idx >= total) return;
    int k = (int)(idx % K);
    int m = (int)(idx / K);
    int g = m / d, v = m % d;
    int c = k / d, u = k % d;
    float acc = 0.f;
    const float* wr = w + ((long long)c * co + g) * 72;
    const float* dc = D + off + u * d + v;
    for (int n = 0; n < 72; ++n)
        acc += dc[n * SO3] * wr[n];
    A[idx] = f2bf(acc * inv);
}

// BT[(b*d+m)*K + c*d+u] = h[(b*ci+c)*455 + off + u*d + m]  (transposed pack)
__global__ void k_packBT(const float* __restrict__ hin, unsigned short* __restrict__ BT,
                         int ci, int d, int off) {
    int N = NB * d;
    int K = ci * d;
    int idx = blockIdx.x * blockDim.x + threadIdx.x;
    if (idx >= K * N) return;
    int k = idx % K, n = idx / K;
    int c = k / d, u = k % d;
    int b = n / d, m = n % d;
    BT[idx] = f2bf(hin[((long long)b * ci + c) * SO3 + off + u * d + m]);
}

// Generic bf16 WMMA GEMM: C[MxN] f32 = A[MxK] x BT[NxK]^T, row-major.
// M % 64 == 0, K % 32 == 0, N % 32 == 0. Block = 256 thr, 64x64 C tile.
__global__ __launch_bounds__(256) void k_gemm(const unsigned short* __restrict__ A,
                                              const unsigned short* __restrict__ BT,
                                              float* __restrict__ C,
                                              int M, int N, int K) {
    __shared__ unsigned short sA[64][32];
    __shared__ unsigned short sBT[64][32];  // [col][k]
    const int tid = threadIdx.x, lane = tid & 31, wv = tid >> 5;
    const int wr = wv >> 1, wc = wv & 1;
    const int la = lane & 15, lg = lane >> 4;
    const int rowbase = blockIdx.y * 64;
    const int colbase = blockIdx.x * 64;
    const bool inW = (colbase + 32 * wc) < N;
    const int sr = (tid * 8) >> 5;        // staging row (0..63)
    const int sk = (tid * 8) & 31;        // staging col (0,8,16,24)
    v8f acc0 = {}, acc1 = {};
    for (int k0 = 0; k0 < K; k0 += 32) {
        __syncthreads();
        {
            *(uint4*)&sA[sr][sk] =
                *(const uint4*)&A[(long long)(rowbase + sr) * K + k0 + sk];
            uint4 bz = {0u, 0u, 0u, 0u};
            *(uint4*)&sBT[sr][sk] = (colbase + sr < N)
                ? *(const uint4*)&BT[(long long)(colbase + sr) * K + k0 + sk]
                : bz;
        }
        __syncthreads();
        if (inW) {
            const unsigned short* arow = sA[16 * wr + la];
            Frag a;
            ldA(a, &arow[16 * lg], &arow[16 * lg + 8]);
#pragma unroll
            for (int t = 0; t < 2; ++t) {
                const unsigned short* brow = sBT[32 * wc + 16 * t + la];
                Frag b;
                ldB(b, &brow[16 * lg]);
                v8f& acc = t ? acc1 : acc0;
                acc = __builtin_amdgcn_wmma_f32_16x16x32_bf16(false, a.v, false, b.v,
                                                              (short)0, acc, false, false);
            }
        }
    }
    if (inW) {
#pragma unroll
        for (int t = 0; t < 2; ++t) {
            v8f acc = t ? acc1 : acc0;
            int col = colbase + 32 * wc + 16 * t + la;
            if (col < N) {
#pragma unroll
                for (int v = 0; v < 8; ++v)
                    C[(long long)(rowbase + 16 * wr + v + 8 * lg) * N + col] = acc[v];
            }
        }
    }
}

// h_out[(b*co+g)*455 + off + v*d + m] = C[(g*d+v)*N + b*d + m]
__global__ void k_unpack(const float* __restrict__ C, float* __restrict__ hout,
                         int co, int d, int off) {
    int N = NB * d;
    long long total = (long long)co * d * N;
    long long idx = (long long)blockIdx.x * blockDim.x + threadIdx.x;
    if (idx >= total) return;
    int n = (int)(idx % N);
    int mrow = (int)(idx / N);
    int g = mrow / d, v = mrow % d;
    int b = n / d, m = n % d;
    hout[((long long)b * co + g) * SO3 + off + v * d + m] = C[idx];
}

// out[b,g,l2+m] = (1/sqrt(256*d)) sum_{c,u} w_lin[c,g,l2+u] * h[(b*256+c)*455+off+u*d+m]
__global__ void k_so3_s2(const float* __restrict__ hin, const float* __restrict__ wlin,
                         float* __restrict__ out, int d, int off, int l2, float inv) {
    int idx = blockIdx.x * blockDim.x + threadIdx.x;
    if (idx >= NB * 128 * d) return;
    int m = idx % d;
    int t = idx / d;
    int g = t % 128;
    int b = t / 128;
    float acc = 0.f;
    for (int c = 0; c < 256; ++c) {
        const float* wr = wlin + ((long long)c * 128 + g) * S2 + l2;
        const float* hr = hin + ((long long)b * 256 + c) * SO3 + off + m;
        for (int u = 0; u < d; ++u)
            acc += wr[u] * hr[u * d];
    }
    out[((long long)b * 128 + g) * S2 + l2 + m] = acc * inv;
}

// ---------------------------------------------------------------------------
// Host orchestration
// ---------------------------------------------------------------------------
extern "C" void kernel_launch(void* const* d_in, const int* in_sizes, int n_in,
                              void* d_out, int out_size, void* d_ws, size_t ws_size,
                              hipStream_t stream) {
    (void)in_sizes; (void)n_in; (void)out_size; (void)ws_size;
    const float* x    = (const float*)d_in[0];
    const float* w_s2 = (const float*)d_in[1];
    const float* Y    = (const float*)d_in[2];
    const float* w1   = (const float*)d_in[3];
    const float* D1   = (const float*)d_in[4];
    const float* w2   = (const float*)d_in[5];
    const float* D2   = (const float*)d_in[6];
    const float* wlin = (const float*)d_in[7];
    const float* Dg   = (const float*)d_in[8];
    float* out = (float*)d_out;

    char* ws = (char*)d_ws;
    size_t o = 0;
    auto alloc = [&](size_t bytes) -> void* {
        void* p = ws + o;
        o = (o + bytes + 255) & ~(size_t)255;
        return p;
    };
    float*          psi0 = (float*)alloc((size_t)128 * 128 * S2 * 4);
    float*          hA   = (float*)alloc((size_t)8192 * SO3 * 4);
    float*          hB   = (float*)alloc((size_t)8192 * SO3 * 4);
    unsigned short* Abf  = (unsigned short*)alloc((size_t)3328 * 3328 * 2);
    unsigned short* Bbf  = (unsigned short*)alloc((size_t)3328 * 416 * 2);
    float*          Cf   = (float*)alloc((size_t)3328 * 416 * 4);
    unsigned short* DgNK = (unsigned short*)alloc((size_t)NGRID * 512 * 2);
    unsigned short* DgT  = (unsigned short*)alloc((size_t)512 * NGRID * 2);

    static const int dl[7]  = {1, 3, 5, 7, 9, 11, 13};
    static const int off[7] = {0, 1, 10, 35, 84, 165, 286};
    static const int l2t[7] = {0, 1, 4, 9, 16, 25, 36};

    // one-time Dg bf16 layouts + psi0 + s2 -> so3 (hA rows = b*128+c, 4096 rows)
    k_prep_dg<<<(NGRID * 512 + 255) / 256, 256, 0, stream>>>(Dg, DgNK, DgT);
    k_psi0<<<(128 * 128 * S2 + 255) / 256, 256, 0, stream>>>(Y, w_s2, psi0);
    for (int l = 0; l < 7; ++l) {
        int d = dl[l];
        int n = NB * 128 * d * d;
        k_s2_so3<<<(n + 255) / 256, 256, 0, stream>>>(x, psi0, hA, d, off[l], l2t[l]);
    }
    // act 1: 4096 rows
    k_act<<<4096 / 16, 256, 0, stream>>>(hA, DgNK, DgT, hB);

    // layer 1: ci=128 -> co=256, hB -> hA
    for (int l = 0; l < 7; ++l) {
        int d = dl[l];
        int M = 256 * d, K = 128 * d, N = NB * d;
        float inv = 1.0f / (sqrtf(72.0f) * sqrtf(128.0f * (float)d));
        k_psi_l<<<(int)(((long long)M * K + 255) / 256), 256, 0, stream>>>(D1, w1, Abf, 128, 256, d, off[l], inv);
        k_packBT<<<(K * N + 255) / 256, 256, 0, stream>>>(hB, Bbf, 128, d, off[l]);
        dim3 g((N + 63) / 64, M / 64);
        k_gemm<<<g, 256, 0, stream>>>(Abf, Bbf, Cf, M, N, K);
        k_unpack<<<(int)(((long long)M * N + 255) / 256), 256, 0, stream>>>(Cf, hA, 256, d, off[l]);
    }
    // act 2: 8192 rows
    k_act<<<8192 / 16, 256, 0, stream>>>(hA, DgNK, DgT, hB);

    // layer 2: ci=256 -> co=256, hB -> hA
    for (int l = 0; l < 7; ++l) {
        int d = dl[l];
        int M = 256 * d, K = 256 * d, N = NB * d;
        float inv = 1.0f / (sqrtf(72.0f) * sqrtf(256.0f * (float)d));
        k_psi_l<<<(int)(((long long)M * K + 255) / 256), 256, 0, stream>>>(D2, w2, Abf, 256, 256, d, off[l], inv);
        k_packBT<<<(K * N + 255) / 256, 256, 0, stream>>>(hB, Bbf, 256, d, off[l]);
        dim3 g((N + 63) / 64, M / 64);
        k_gemm<<<g, 256, 0, stream>>>(Abf, Bbf, Cf, M, N, K);
        k_unpack<<<(int)(((long long)M * N + 255) / 256), 256, 0, stream>>>(Cf, hA, 256, d, off[l]);
    }
    // act 3: 8192 rows
    k_act<<<8192 / 16, 256, 0, stream>>>(hA, DgNK, DgT, hB);

    // final so3 -> s2 projection
    for (int l = 0; l < 7; ++l) {
        int d = dl[l];
        int n = NB * 128 * d;
        float inv = 1.0f / sqrtf(256.0f * (float)d);
        k_so3_s2<<<(n + 255) / 256, 256, 0, stream>>>(hB, wlin, out, d, off[l], l2t[l], inv);
    }
}
